// VSMLMV4_1_6545530159529
// MI455X (gfx1250) — matrix operand order repaired and checked by reading, not compiled, not run on hardware
//
#include <hip/hip_runtime.h>
#include <math.h>
#include <stdint.h>

// ---------------------------------------------------------------------------
// Static problem config (matches reference)
// ---------------------------------------------------------------------------
#define BB 4
#define LL 2048
#define DD 512
#define DRR 256
#define RR 3
#define PP 5
#define PHH 3
#define FF 1536
#define NTOK (BB * LL)              // 8192
#define RDR (RR * DRR)              // 768
#define GIN (RDR + DD)              // 1280

typedef __attribute__((ext_vector_type(16))) __bf16 v16bf;
typedef __attribute__((ext_vector_type(8)))  __bf16 v8bf;
typedef __attribute__((ext_vector_type(8)))  float  v8f;
typedef __attribute__((ext_vector_type(4)))  unsigned int   v4u;
typedef __attribute__((ext_vector_type(4)))  unsigned short v4us;

__device__ __forceinline__ unsigned short f2bf(float f) {
    union { float f; unsigned u; } c; c.f = f;
    unsigned r = c.u + 0x7FFFu + ((c.u >> 16) & 1u);   // round-to-nearest-even
    return (unsigned short)(r >> 16);
}
__device__ __forceinline__ float sigmoidf_(float x) { return 1.0f / (1.0f + expf(-x)); }
__device__ __forceinline__ float gelu_exact(float x) {
    return 0.5f * x * (1.0f + erff(x * 0.70710678118654752440f));
}

// Block-wide sum (blockDim.x == 256), fixed order -> deterministic.
__device__ __forceinline__ float block_reduce_sum256(float v, float* red) {
    int t = threadIdx.x;
    red[t] = v; __syncthreads();
    for (int o = 128; o > 0; o >>= 1) {
        if (t < o) red[t] += red[t + o];
        __syncthreads();
    }
    float r = red[0]; __syncthreads();
    return r;
}

// Per-token LayerNorm: 256 threads, each owning cols t and t+256 of a 512 row.
__device__ __forceinline__ void ln_token(const float* __restrict__ xr,
                                         const float* __restrict__ s,
                                         const float* __restrict__ b,
                                         float* red, float& xn0, float& xn1) {
    int t = threadIdx.x;
    float a0 = xr[t], a1 = xr[t + 256];
    float mean = block_reduce_sum256(a0 + a1, red) * (1.0f / 512.0f);
    float d0 = a0 - mean, d1 = a1 - mean;
    float var = block_reduce_sum256(d0 * d0 + d1 * d1, red) * (1.0f / 512.0f);
    float inv = rsqrtf(var + 1e-5f);
    xn0 = d0 * inv * s[t]       + b[t];
    xn1 = d1 * inv * s[t + 256] + b[t + 256];
}

// ---------------------------------------------------------------------------
// bf16 WMMA GEMM:  C[M,N] = act(A[M,K] @ B[K,N] + bias)
// Block: 1024 thr (32 waves, 8 M-rows x 4 N-cols) -> 128(M) x 256(N) tile.
// Each wave: 16 rows x 64 cols = 4 f32 accumulators, K-step 64 (2 WMMA/acc).
// A tile staged via global_load_async_to_lds_b128 (ASYNCcnt-tracked DMA).
// B tile staged TRANSPOSED in LDS (row stride 72 ushorts = 144 B) so every
// fragment read is a contiguous ds_load_b128. A scheduling fence between the
// fragment-load block and the WMMA block keeps all 20 b128 loads issued
// up-front so dscnt waits become incremental instead of full drains.
// ---------------------------------------------------------------------------
#define KSTEP 64
#define LSTR  72      // LDS row stride in ushorts (64 data + 8 pad)

template <int ACT, int OUTF32, int N, int K>
__global__ __launch_bounds__(1024)
void gemm_bf16_wmma(const unsigned short* __restrict__ A,
                    const unsigned short* __restrict__ Bm,
                    const float* __restrict__ bias,
                    void* __restrict__ Cout) {
    __shared__ unsigned short As[128 * LSTR];   // row-major   [m][k]
    __shared__ unsigned short Bt[256 * LSTR];   // transposed  [n][k]

    const int t    = threadIdx.x;
    const int wave = t >> 5, lane = t & 31;
    const int half = lane >> 4, mr = lane & 15;
    const int mrow = wave >> 2, ncol = wave & 3;
    const int m0 = blockIdx.y * 128;
    const int n0 = blockIdx.x * 256;

    // staging coordinates
    const int ar  = t >> 3;                 // A row 0..127
    const int ac  = (t & 7) * 8;            // A col 0..56 (step 8)
    const int bc  = t & 255;                // B col 0..255
    const int bkq = (t >> 8) * 16;          // B k base {0,16,32,48}

    const unsigned a_lds = (unsigned)(uintptr_t)(&As[ar * LSTR + ac]);
    unsigned long long ga =
        (unsigned long long)(uintptr_t)(A + (size_t)(m0 + ar) * K + ac);
    const unsigned short* bptr = Bm + (size_t)bkq * N + (n0 + bc);

    const int arow  = (mrow * 16 + mr) * LSTR + half * 16;
    const int bbase = (ncol * 64 + mr) * LSTR + half * 16;

    v8f acc[4] = {};

    for (int k0 = 0; k0 < K; k0 += KSTEP) {
        // ---- A tile 128x64: async DMA, 16 B per lane, no VGPR round-trip ----
        asm volatile("global_load_async_to_lds_b128 %0, %1, off"
                     :: "v"(a_lds), "v"(ga) : "memory");
        ga += (unsigned long long)(KSTEP * 2);

        // ---- B tile 64x256 -> transposed LDS: coalesced 2B/lane global reads,
        //      packed b64 LDS stores ----
        #pragma unroll
        for (int j = 0; j < 4; ++j) {
            v4us pk;
            #pragma unroll
            for (int i = 0; i < 4; ++i)
                pk[i] = bptr[(size_t)(4 * j + i) * N];
            *(v4us*)(&Bt[bc * LSTR + bkq + 4 * j]) = pk;
        }
        bptr += (size_t)KSTEP * N;
        if (k0 + KSTEP < K)
            __builtin_prefetch(bptr, 0, 1);

        asm volatile("s_wait_asynccnt 0x0" ::: "memory");
        __syncthreads();

        // ---- issue ALL fragment loads for this K-tile, in consumption order --
        v16bf af0, af1, bf0[4], bf1[4];
        {
            v8bf lo = *(const v8bf*)(&As[arow]);
            v8bf hi = *(const v8bf*)(&As[arow + 8]);
            af0 = __builtin_shufflevector(lo, hi,
                    0, 1, 2, 3, 4, 5, 6, 7, 8, 9, 10, 11, 12, 13, 14, 15);
        }
        #pragma unroll
        for (int n = 0; n < 4; ++n)
            bf0[n] = *(const v16bf*)(&Bt[bbase + n * 16 * LSTR]);
        {
            v8bf lo = *(const v8bf*)(&As[arow + 32]);
            v8bf hi = *(const v8bf*)(&As[arow + 40]);
            af1 = __builtin_shufflevector(lo, hi,
                    0, 1, 2, 3, 4, 5, 6, 7, 8, 9, 10, 11, 12, 13, 14, 15);
        }
        #pragma unroll
        for (int n = 0; n < 4; ++n)
            bf1[n] = *(const v16bf*)(&Bt[bbase + n * 16 * LSTR + 32]);

        // Fence: DS loads can't sink below; WMMAs depend on the loaded vregs,
        // so waits become incremental (s_wait_dscnt N) instead of full drains.
        asm volatile("" ::: "memory");

        // ---- 8 back-to-back WMMAs (acc reuse distance = 4) ----
        #pragma unroll
        for (int n = 0; n < 4; ++n)
            acc[n] = __builtin_amdgcn_wmma_f32_16x16x32_bf16(
                false, af0, false, bf0[n], (short)0, acc[n], false, false);
        #pragma unroll
        for (int n = 0; n < 4; ++n)
            acc[n] = __builtin_amdgcn_wmma_f32_16x16x32_bf16(
                false, af1, false, bf1[n], (short)0, acc[n], false, false);
        __syncthreads();
    }

    // Epilogue: C layout -> VGPR r holds row r (+8 for hi half), col = lane%16
    #pragma unroll
    for (int n = 0; n < 4; ++n) {
        int col = n0 + ncol * 64 + n * 16 + mr;
        float bv = bias[col];
        #pragma unroll
        for (int r = 0; r < 8; ++r) {
            int row = m0 + mrow * 16 + r + half * 8;
            float v = acc[n][r] + bv;
            if (ACT) v = gelu_exact(v);
            if (OUTF32) ((float*)Cout)[(size_t)row * N + col] = v;
            else ((unsigned short*)Cout)[(size_t)row * N + col] = f2bf(v);
        }
    }
}

// ---------------------------------------------------------------------------
// Small kernels
// ---------------------------------------------------------------------------
__global__ void cvt_bf16_kernel(const float* __restrict__ s,
                                unsigned short* __restrict__ d, int n) {
    int i = blockIdx.x * 256 + threadIdx.x;
    if (i < n) d[i] = f2bf(s[i]);
}

// pass_w = sigmoid(tile(reg_init.flat, P) @ ms3_w + ms3_b)
__global__ void ms3_kernel(const float* __restrict__ reg_init,
                           const float* __restrict__ w,
                           const float* __restrict__ b,
                           float* __restrict__ pw) {
    int j = threadIdx.x;
    if (j >= PP) return;
    float s = b[j];
    for (int i = 0; i < PP * RDR; ++i) s += reg_init[i % RDR] * w[i * PP + j];
    pw[j] = sigmoidf_(s);
}

// qk = Kmat @ (banks_flat[0:nb] @ Qmat)  -- single block of 512 threads
__global__ void qk_kernel(const float* __restrict__ banks, int nb,
                          const float* __restrict__ Qmat,
                          const float* __restrict__ Kmat,
                          float* __restrict__ qk) {
    __shared__ float qs[DD];
    int d = threadIdx.x;
    float q = 0.0f;
    for (int i = 0; i < nb; ++i) q += banks[i] * Qmat[(size_t)i * DD + d];
    qs[d] = q; __syncthreads();
    const float* row = Kmat + (size_t)d * DD;
    float s = 0.0f;
    for (int dp = 0; dp < DD; ++dp) s += row[dp] * qs[dp];
    qk[d] = s;
}

// score[tok] = (LN(resid[tok]) . qk) * scale   -- one block per token
__global__ __launch_bounds__(256)
void score_kernel(const float* __restrict__ resid,
                  const float* __restrict__ s, const float* __restrict__ b,
                  const float* __restrict__ qk, float* __restrict__ score,
                  float scale) {
    __shared__ float red[256];
    int tok = blockIdx.x, t = threadIdx.x;
    float xn0, xn1;
    ln_token(resid + (size_t)tok * DD, s, b, red, xn0, xn1);
    float dot = block_reduce_sum256(xn0 * qk[t] + xn1 * qk[t + 256], red);
    if (t == 0) score[tok] = dot * scale;
}

// in-place softmax over L per batch -- one block per batch
__global__ __launch_bounds__(256)
void softmax_kernel(float* __restrict__ sc) {
    __shared__ float red[256];
    float* s = sc + (size_t)blockIdx.x * LL;
    int t = threadIdx.x;
    float mx = -3.4e38f;
    for (int l = t; l < LL; l += 256) mx = fmaxf(mx, s[l]);
    red[t] = mx; __syncthreads();
    for (int o = 128; o > 0; o >>= 1) { if (t < o) red[t] = fmaxf(red[t], red[t + o]); __syncthreads(); }
    mx = red[0]; __syncthreads();
    float sum = 0.0f;
    for (int l = t; l < LL; l += 256) { float e = expf(s[l] - mx); s[l] = e; sum += e; }
    sum = block_reduce_sum256(sum, red);
    float inv = 1.0f / sum;
    for (int l = t; l < LL; l += 256) s[l] *= inv;
}

// partial[blk] = sum over 64 tokens of a[tok] * LN(resid[tok])   (128 blocks)
__global__ __launch_bounds__(256)
void wsum_partial_kernel(const float* __restrict__ resid,
                         const float* __restrict__ s, const float* __restrict__ b,
                         const float* __restrict__ attw,
                         float* __restrict__ partial) {
    __shared__ float red[256];
    int blk = blockIdx.x, t = threadIdx.x;
    int bb = blk >> 5, c = blk & 31;
    float a0 = 0.0f, a1 = 0.0f;
    for (int j = 0; j < 64; ++j) {
        int tok = bb * LL + c * 64 + j;
        float xn0, xn1;
        ln_token(resid + (size_t)tok * DD, s, b, red, xn0, xn1);
        float w = attw[tok];
        a0 += w * xn0; a1 += w * xn1;
    }
    partial[(size_t)blk * DD + t]       = a0;
    partial[(size_t)blk * DD + 256 + t] = a1;
}

// s4: mean over (b, chunks): out[512] = sum(all 128 partials) / B
__global__ void wsum_reduce_mean(const float* __restrict__ partial, float* __restrict__ out) {
    int d = threadIdx.x;
    float s = 0.0f;
    for (int i = 0; i < BB * 32; ++i) s += partial[(size_t)i * DD + d];
    out[d] = s * (1.0f / (float)BB);
}

// m4: per-batch: out[b,512] = sum over 32 chunks
__global__ void wsum_reduce_b(const float* __restrict__ partial, float* __restrict__ out) {
    int b = blockIdx.x, d = threadIdx.x;
    float s = 0.0f;
    for (int i = 0; i < 32; ++i) s += partial[(size_t)(b * 32 + i) * DD + d];
    out[(size_t)b * DD + d] = s;
}

// regs = reg_init + ((wsumMean @ s4_v) @ s4_sum)   -- one block, 768 threads
__global__ __launch_bounds__(768)
void s4_final_kernel(const float* __restrict__ wsumMean,
                     const float* __restrict__ s4_v,
                     const float* __restrict__ s4_sum,
                     const float* __restrict__ reg_init,
                     float* __restrict__ regs) {
    __shared__ float summ[DD];
    int t = threadIdx.x;
    if (t < DD) {
        float s = 0.0f;
        for (int d = 0; d < DD; ++d) s += wsumMean[d] * s4_v[(size_t)d * DD + t];
        summ[t] = s;
    }
    __syncthreads();
    float s = 0.0f;
    for (int d = 0; d < DD; ++d) s += summ[d] * s4_sum[(size_t)d * RDR + t];
    regs[t] = reg_init[t] + s;
}

// LN -> bf16 activations for the FFN GEMM
__global__ __launch_bounds__(256)
void ln_store_bf16(const float* __restrict__ resid,
                   const float* __restrict__ s, const float* __restrict__ b,
                   unsigned short* __restrict__ out) {
    __shared__ float red[256];
    int tok = blockIdx.x, t = threadIdx.x;
    float xn0, xn1;
    ln_token(resid + (size_t)tok * DD, s, b, red, xn0, xn1);
    out[(size_t)tok * DD + t]       = f2bf(xn0);
    out[(size_t)tok * DD + 256 + t] = f2bf(xn1);
}

// final LN -> f32 output
__global__ __launch_bounds__(256)
void ln_store_f32(const float* __restrict__ resid,
                  const float* __restrict__ s, const float* __restrict__ b,
                  float* __restrict__ out) {
    __shared__ float red[256];
    int tok = blockIdx.x, t = threadIdx.x;
    float xn0, xn1;
    ln_token(resid + (size_t)tok * DD, s, b, red, xn0, xn1);
    out[(size_t)tok * DD + t]       = xn0;
    out[(size_t)tok * DD + 256 + t] = xn1;
}

// column partial sums of delta (64 blocks x 128 rows)
__global__ __launch_bounds__(256)
void colsum_partial_kernel(const float* __restrict__ delta, float* __restrict__ partial) {
    int blk = blockIdx.x, t = threadIdx.x;
    float s0 = 0.0f, s1 = 0.0f;
    for (int r = 0; r < 128; ++r) {
        const float* row = delta + (size_t)(blk * 128 + r) * DD;
        s0 += row[t]; s1 += row[t + 256];
    }
    partial[(size_t)blk * DD + t]       = s0;
    partial[(size_t)blk * DD + 256 + t] = s1;
}

// gate / write-gate / register update for one (p, ph)   -- one block, 768 thr
__global__ __launch_bounds__(768)
void phase_post_kernel(const float* __restrict__ colsum,
                       float* __restrict__ regs,
                       const float* __restrict__ gw, const float* __restrict__ gb,
                       const float* __restrict__ wgw, const float* __restrict__ wgb,
                       const float* __restrict__ wpw,
                       const float* __restrict__ passw, int p,
                       float* __restrict__ gvec) {
    __shared__ float summary[DD];
    int t = threadIdx.x;
    if (t < DD) {
        float s = 0.0f;
        for (int i = 0; i < 64; ++i) s += colsum[(size_t)i * DD + t];
        summary[t] = s * (1.0f / (float)NTOK);
    }
    __syncthreads();
    float gsum = 0.0f;
    if (t < DD) {
        gsum = gb[t];
        for (int i = 0; i < RDR; ++i) gsum += regs[i] * gw[(size_t)i * DD + t];
        for (int i = 0; i < DD;  ++i) gsum += summary[i] * gw[(size_t)(RDR + i) * DD + t];
    }
    int r = t >> 8, k = t & 255;
    float wgs = wgb[r];
    float up = 0.0f;
    for (int d = 0; d < DD; ++d) {
        float sd = summary[d];
        wgs += sd * wgw[(size_t)r * DD + d];
        up  += sd * wpw[((size_t)(r * DD + d)) * DRR + k];
    }
    __syncthreads();                       // all reads of regs[] complete
    if (t < DD) gvec[t] = passw[p] * sigmoidf_(gsum);
    regs[t] += sigmoidf_(wgs) * up;
}

__global__ __launch_bounds__(256)
void resid_update_kernel(float* __restrict__ resid,
                         const float* __restrict__ delta,
                         const float* __restrict__ gvec) {
    size_t i = (size_t)blockIdx.x * 256 + threadIdx.x;
    resid[i] += gvec[i & (DD - 1)] * delta[i];
}

__global__ void bank_copy_kernel(const float* __restrict__ regs, float* __restrict__ bank) {
    bank[threadIdx.x] = regs[threadIdx.x];
}

// summ[b] = wsum[b] @ m4_v ; add[b] = summ[b] @ m4_out   -- one block, 512 thr
__global__ __launch_bounds__(512)
void m4_final_kernel(const float* __restrict__ wsumB,
                     const float* __restrict__ m4_v,
                     const float* __restrict__ m4_out,
                     float* __restrict__ add) {
    __shared__ float summ[DD];
    int t = threadIdx.x;
    for (int b = 0; b < BB; ++b) {
        const float* w = wsumB + (size_t)b * DD;
        float s = 0.0f;
        for (int d = 0; d < DD; ++d) s += w[d] * m4_v[(size_t)d * DD + t];
        summ[t] = s; __syncthreads();
        float o = 0.0f;
        for (int d = 0; d < DD; ++d) o += summ[d] * m4_out[(size_t)d * DD + t];
        add[(size_t)b * DD + t] = o; __syncthreads();
    }
}

__global__ __launch_bounds__(256)
void resid_add_b_kernel(float* __restrict__ resid, const float* __restrict__ add) {
    size_t i = (size_t)blockIdx.x * 256 + threadIdx.x;
    int b = (int)(i >> 20);                 // L*D = 2^20
    resid[i] += add[(size_t)b * DD + (i & (DD - 1))];
}

// ---------------------------------------------------------------------------
// Host orchestration
// ---------------------------------------------------------------------------
extern "C" void kernel_launch(void* const* d_in, const int* in_sizes, int n_in,
                              void* d_out, int out_size, void* d_ws, size_t ws_size,
                              hipStream_t stream) {
    (void)in_sizes; (void)n_in; (void)out_size; (void)ws_size;

    const float* x        = (const float*)d_in[0];
    const float* reg_init = (const float*)d_in[1];
    const float* ffn_ln_s = (const float*)d_in[2];
    const float* ffn_ln_b = (const float*)d_in[3];
    const float* ffn_w1   = (const float*)d_in[4];
    const float* ffn_b1   = (const float*)d_in[5];
    const float* ffn_w2   = (const float*)d_in[6];
    const float* ffn_b2   = (const float*)d_in[7];
    const float* gate_w   = (const float*)d_in[8];
    const float* gate_b   = (const float*)d_in[9];
    const float* wproj_w  = (const float*)d_in[10];
    const float* wgate_w  = (const float*)d_in[11];
    const float* wgate_b  = (const float*)d_in[12];
    const float* s4_q     = (const float*)d_in[13];
    const float* s4_k     = (const float*)d_in[14];
    const float* s4_v     = (const float*)d_in[15];
    const float* s4_sum   = (const float*)d_in[16];
    const float* s4_ln_s  = (const float*)d_in[17];
    const float* s4_ln_b  = (const float*)d_in[18];
    const float* ms3_w    = (const float*)d_in[19];
    const float* ms3_b    = (const float*)d_in[20];
    const float* m4_q     = (const float*)d_in[21];
    const float* m4_k     = (const float*)d_in[22];
    const float* m4_v     = (const float*)d_in[23];
    const float* m4_out   = (const float*)d_in[24];
    const float* m4_ln_s  = (const float*)d_in[25];
    const float* m4_ln_b  = (const float*)d_in[26];
    const float* out_ln_s = (const float*)d_in[27];
    const float* out_ln_b = (const float*)d_in[28];

    // -------- workspace layout --------
    char* ws = (char*)d_ws;
    size_t off = 0;
    auto alloc = [&](size_t bytes) { char* p = ws + off; off += (bytes + 255) & ~(size_t)255; return p; };
    float*          resid  = (float*)         alloc((size_t)NTOK * DD * 4);        // 16.8 MB
    unsigned short* xnb    = (unsigned short*)alloc((size_t)NTOK * DD * 2);        //  8.4 MB
    unsigned short* hbuf   = (unsigned short*)alloc((size_t)NTOK * FF * 2);        // 25.2 MB
    float*          delta  = (float*)         alloc((size_t)NTOK * DD * 4);        // 16.8 MB
    unsigned short* w1b    = (unsigned short*)alloc((size_t)PHH * DD * FF * 2);    //  4.7 MB
    unsigned short* w2b    = (unsigned short*)alloc((size_t)PHH * FF * DD * 2);    //  4.7 MB
    float*          qk     = (float*)         alloc(DD * 4);
    float*          score  = (float*)         alloc((size_t)NTOK * 4);
    float*          wpart  = (float*)         alloc((size_t)BB * 32 * DD * 4);
    float*          wsum   = (float*)         alloc((size_t)BB * DD * 4);
    float*          colsum = (float*)         alloc((size_t)64 * DD * 4);
    float*          gvec   = (float*)         alloc(DD * 4);
    float*          regs   = (float*)         alloc(RDR * 4);
    float*          banks  = (float*)         alloc((size_t)PP * RDR * 4);
    float*          passw  = (float*)         alloc(PP * 4);
    float*          addv   = (float*)         alloc((size_t)BB * DD * 4);

    const float scale = 1.0f / sqrtf((float)DD);
    const size_t elems = (size_t)NTOK * DD;   // 4,194,304

    // residual = x
    hipMemcpyAsync(resid, x, elems * sizeof(float), hipMemcpyDeviceToDevice, stream);

    // weights -> bf16 once
    {
        int n1 = PHH * DD * FF;
        cvt_bf16_kernel<<<(n1 + 255) / 256, 256, 0, stream>>>(ffn_w1, w1b, n1);
        cvt_bf16_kernel<<<(n1 + 255) / 256, 256, 0, stream>>>(ffn_w2, w2b, n1);
    }

    // pass gates
    ms3_kernel<<<1, 32, 0, stream>>>(reg_init, ms3_w, ms3_b, passw);

    for (int p = 0; p < PP; ++p) {
        // ---- S4Intelligence read (collapsed single-query attention) ----
        qk_kernel<<<1, DD, 0, stream>>>(banks, p * RDR, s4_q, s4_k, qk);
        score_kernel<<<NTOK, 256, 0, stream>>>(resid, s4_ln_s, s4_ln_b, qk, score, scale);
        softmax_kernel<<<BB, 256, 0, stream>>>(score);
        wsum_partial_kernel<<<BB * 32, 256, 0, stream>>>(resid, s4_ln_s, s4_ln_b, score, wpart);
        wsum_reduce_mean<<<1, DD, 0, stream>>>(wpart, wsum);
        s4_final_kernel<<<1, RDR, 0, stream>>>(wsum, s4_v, s4_sum, reg_init, regs);

        // ---- three FFN phases ----
        for (int ph = 0; ph < PHH; ++ph) {
            ln_store_bf16<<<NTOK, 256, 0, stream>>>(resid, ffn_ln_s + ph * DD,
                                                    ffn_ln_b + ph * DD, xnb);
            // h = gelu(xn @ w1 + b1)     [8192,512] x [512,1536] -> bf16
            gemm_bf16_wmma<1, 0, FF, DD><<<dim3(FF / 256, NTOK / 128), 1024, 0, stream>>>(
                xnb, w1b + (size_t)ph * DD * FF, ffn_b1 + ph * FF, (void*)hbuf);
            // delta = h @ w2 + b2        [8192,1536] x [1536,512] -> f32
            gemm_bf16_wmma<0, 1, DD, FF><<<dim3(DD / 256, NTOK / 128), 1024, 0, stream>>>(
                hbuf, w2b + (size_t)ph * FF * DD, ffn_b2 + ph * DD, (void*)delta);

            colsum_partial_kernel<<<64, 256, 0, stream>>>(delta, colsum);
            size_t pp = (size_t)(p * PHH + ph);
            phase_post_kernel<<<1, RDR, 0, stream>>>(
                colsum, regs,
                gate_w + pp * GIN * DD, gate_b + pp * DD,
                wgate_w + pp * RR * DD, wgate_b + pp * RR,
                wproj_w + pp * RR * DD * DRR, passw, p, gvec);
            resid_update_kernel<<<(unsigned)(elems / 256), 256, 0, stream>>>(resid, delta, gvec);
        }
        bank_copy_kernel<<<1, RDR, 0, stream>>>(regs, banks + (size_t)p * RDR);
    }

    // ---- MetaS4 final scan ----
    qk_kernel<<<1, DD, 0, stream>>>(banks, PP * RDR, m4_q, m4_k, qk);
    score_kernel<<<NTOK, 256, 0, stream>>>(resid, m4_ln_s, m4_ln_b, qk, score, scale);
    softmax_kernel<<<BB, 256, 0, stream>>>(score);
    wsum_partial_kernel<<<BB * 32, 256, 0, stream>>>(resid, m4_ln_s, m4_ln_b, score, wpart);
    wsum_reduce_b<<<BB, DD, 0, stream>>>(wpart, wsum);
    m4_final_kernel<<<1, DD, 0, stream>>>(wsum, m4_v, m4_out, addv);
    resid_add_b_kernel<<<(unsigned)(elems / 256), 256, 0, stream>>>(resid, addv);

    // ---- output LN ----
    ln_store_f32<<<NTOK, 256, 0, stream>>>(resid, out_ln_s, out_ln_b, (float*)d_out);
}
